// PointNetSetAbstraction_49237505082101
// MI455X (gfx1250) — matrix operand order repaired and checked by reading, not compiled, not run on hardware
//
#include <hip/hip_runtime.h>

// Problem constants (match reference)
#define BB   8
#define NN   8192
#define SS   1024
#define KK   32
#define DD   64
#define CIN  67      // 3 + D
#define CPAD 96      // padded to 3 k-steps of 32
#define R2   0.04f   // RADIUS^2

typedef __attribute__((ext_vector_type(16))) _Float16 v16h;
typedef __attribute__((ext_vector_type(8)))  float    v8f;

union FragU { v16h v; unsigned int u[8]; };

// A-matrix fragment (16x32 f16, M x K). CDNA5 layout:
// lane = 16*half + m; VGPR v (v<4): K = 2v + 8*half ; (v>=4): K = 16 + 2(v-4) + 8*half
__device__ __forceinline__ v16h frag_a(const _Float16* base, int stride, int kbase, int lane) {
  const int m = lane & 15, hh = lane >> 4;
  FragU r;
#pragma unroll
  for (int v = 0; v < 8; ++v) {
    const int k = kbase + 2 * v + 8 * hh + ((v >= 4) ? 8 : 0);
    r.u[v] = *reinterpret_cast<const unsigned int*>(base + m * stride + k);
  }
  return r.v;
}

// B-matrix fragment (32x16 f16, K x N), B[k][n] = w[n][k] with w stored row-major [o][c].
// lane = 16*half + n; VGPR v (v<4): K = 2v + 8*half ; (v>=4): K = 16 + 2(v-4) + 8*half  (pairs contiguous in c => one b32 DS read)
__device__ __forceinline__ v16h frag_b(const _Float16* wbase, int stride, int kbase, int row0, int lane) {
  const int n = lane & 15, hh = lane >> 4;
  FragU r;
#pragma unroll
  for (int v = 0; v < 8; ++v) {
    const int k = kbase + 2 * v + 8 * hh + ((v >= 4) ? 8 : 0);
    r.u[v] = *reinterpret_cast<const unsigned int*>(wbase + (row0 + n) * stride + k);
  }
  return r.v;
}

__device__ __forceinline__ v8f wmma16(v16h a, v16h b, v8f c) {
  return __builtin_amdgcn_wmma_f32_16x16x32_f16(false, a, false, b, (short)0, c, false, false);
}

// Store a 16x16 f32 D tile (bias + ReLU) back to LDS as f16.
// D layout: lane = 16*half + n ; VGPR g holds row (g + 8*half), col n.
__device__ __forceinline__ void store_h(_Float16* base, int stride, int colbase, int lane, v8f acc, float bias) {
  const int n = lane & 15, hh = lane >> 4;
#pragma unroll
  for (int g = 0; g < 8; ++g) {
    float v = acc[g] + bias;
    v = v > 0.f ? v : 0.f;
    base[(g + 8 * hh) * stride + colbase + n] = (_Float16)v;
  }
}

// ---------------------------------------------------------------------------
// Kernel 0: transpose points (B,D,N) f32 -> ptsT (B,N,D) f16.
// Makes each gathered point one contiguous 128-byte row (one L2 line).
// ---------------------------------------------------------------------------
__global__ __launch_bounds__(256) void transpose_kernel(const float* __restrict__ points,
                                                        _Float16* __restrict__ ptsT) {
  __shared__ _Float16 tile[64][65];  // +1 pad to dodge bank conflicts
  const int b  = (int)blockIdx.x >> 7;          // 128 n-tiles per batch
  const int n0 = ((int)blockIdx.x & 127) * 64;
  const int tid = (int)threadIdx.x;
  for (int i = tid; i < 64 * 64; i += 256) {
    const int c = i >> 6, nr = i & 63;          // consecutive tid -> consecutive n (coalesced read)
    tile[nr][c] = (_Float16)points[((size_t)b * DD + c) * NN + n0 + nr];
  }
  __syncthreads();
  for (int i = tid; i < 64 * 64; i += 256) {
    const int nr = i >> 6, c = i & 63;          // consecutive tid -> consecutive c (coalesced write)
    ptsT[((size_t)b * NN + n0 + nr) * DD + c] = tile[nr][c];
  }
}

// ---------------------------------------------------------------------------
// Kernel 1: farthest point sampling. One block per batch. 1024 threads,
// 8 points per thread held in registers. Writes new_xyz in (B,3,S) layout.
// ---------------------------------------------------------------------------
__global__ __launch_bounds__(1024) void fps_kernel(const float* __restrict__ xyz,
                                                   float* __restrict__ out_xyz) {
  const int b = blockIdx.x;
  const int t = (int)threadIdx.x;
  const int lane = t & 31, wid = t >> 5;
  const float* xb = xyz + (size_t)b * 3 * NN;

  float px[8], py[8], pz[8], dmin[8];
#pragma unroll
  for (int j = 0; j < 8; ++j) {
    const int p = t + j * 1024;
    px[j] = xb[p];
    py[j] = xb[NN + p];
    pz[j] = xb[2 * NN + p];
    dmin[j] = 1e10f;
  }

  __shared__ float wv[32];
  __shared__ int   wi[32];
  __shared__ float lc[3];
  __shared__ int   lidx;
  if (t == 0) { lidx = 0; lc[0] = xb[0]; lc[1] = xb[NN]; lc[2] = xb[2 * NN]; }
  __syncthreads();

  for (int s = 0; s < SS; ++s) {
    const float lx = lc[0], ly = lc[1], lz = lc[2];
    if (t == 0) {
      out_xyz[(b * 3 + 0) * SS + s] = lx;
      out_xyz[(b * 3 + 1) * SS + s] = ly;
      out_xyz[(b * 3 + 2) * SS + s] = lz;
    }
    float bv = -1.f; int bi = 0;
#pragma unroll
    for (int j = 0; j < 8; ++j) {
      const float dx = px[j] - lx, dy = py[j] - ly, dz = pz[j] - lz;
      const float dd = dx * dx + dy * dy + dz * dz;
      dmin[j] = fminf(dmin[j], dd);
      const int pi = t + j * 1024;
      if (dmin[j] > bv || (dmin[j] == bv && pi < bi)) { bv = dmin[j]; bi = pi; }
    }
    // wave32 argmax (first-index tie break, as jnp.argmax)
#pragma unroll
    for (int off = 16; off >= 1; off >>= 1) {
      const float ov = __shfl_xor(bv, off);
      const int   oi = __shfl_xor(bi, off);
      if (ov > bv || (ov == bv && oi < bi)) { bv = ov; bi = oi; }
    }
    if (lane == 0) { wv[wid] = bv; wi[wid] = bi; }
    __syncthreads();
    if (wid == 0) {
      bv = wv[lane]; bi = wi[lane];
#pragma unroll
      for (int off = 16; off >= 1; off >>= 1) {
        const float ov = __shfl_xor(bv, off);
        const int   oi = __shfl_xor(bi, off);
        if (ov > bv || (ov == bv && oi < bi)) { bv = ov; bi = oi; }
      }
      if (lane == 0) {
        lidx = bi;
        lc[0] = xb[bi]; lc[1] = xb[NN + bi]; lc[2] = xb[2 * NN + bi];
      }
    }
    __syncthreads();
  }
  (void)lidx;
}

// ---------------------------------------------------------------------------
// Kernel 2: ball query. One wave per query point; collect the K smallest
// matching indices (== reference top_k over index-candidates), pad with first.
// ---------------------------------------------------------------------------
__global__ __launch_bounds__(256) void ballq_kernel(const float* __restrict__ xyz,
                                                    const float* __restrict__ new_xyz,
                                                    int* __restrict__ idxws) {
  const int tid = (int)threadIdx.x;
  const int lane = tid & 31;
  const int sid = (int)blockIdx.x * 8 + (tid >> 5);
  const int b = sid >> 10, s = sid & 1023;

  const float cx = new_xyz[(b * 3 + 0) * SS + s];
  const float cy = new_xyz[(b * 3 + 1) * SS + s];
  const float cz = new_xyz[(b * 3 + 2) * SS + s];
  const float* xb = xyz + (size_t)b * 3 * NN;
  const int base = sid * KK;

  int cnt = 0;
  int first = -1;
  for (int n0 = 0; n0 < NN && cnt < KK; n0 += 32) {
    const int n = n0 + lane;
    __builtin_prefetch(xb + n + 256, 0, 0);  // global_prefetch_b8
    const float dx = xb[n] - cx;
    const float dy = xb[NN + n] - cy;
    const float dz = xb[2 * NN + n] - cz;
    const bool ok = (dx * dx + dy * dy + dz * dz) <= R2;
    const unsigned int mask = (unsigned int)__ballot(ok);
    if (first < 0 && mask != 0u) first = n0 + __builtin_ctz(mask);
    const int rank = cnt + __popc(mask & ((1u << lane) - 1u));
    if (ok && rank < KK) idxws[base + rank] = n;
    cnt += __popc(mask);
  }
  if (cnt < KK) {
    if (first < 0) first = 0;
    if (lane >= cnt) idxws[base + lane] = first;  // KK == wave size
  }
}

// ---------------------------------------------------------------------------
// Kernel 3: gather + 3-layer MLP (WMMA f16->f32) + max over K + att mean.
// 4 waves per block, one (b,s) group per wave. Weights staged in LDS once.
// Channel order permuted: points -> cols 0..63 (16B-aligned b128 copies),
// g_norm -> cols 64..66; w0 columns permuted identically.
// ---------------------------------------------------------------------------
__global__ __launch_bounds__(128) void mlp_kernel(const float* __restrict__ xyz,
                                                  const float* __restrict__ points,
                                                  const _Float16* __restrict__ ptsT,
                                                  const float* __restrict__ att,
                                                  const float* __restrict__ w0, const float* __restrict__ b0,
                                                  const float* __restrict__ w1, const float* __restrict__ b1,
                                                  const float* __restrict__ w2, const float* __restrict__ b2,
                                                  const int* __restrict__ idxws,
                                                  const float* __restrict__ new_xyz,
                                                  float* __restrict__ out_pts,
                                                  float* __restrict__ out_att) {
  __shared__ __align__(16) _Float16 w0l[64][CPAD];      // 12 KB (columns permuted + zero pad)
  __shared__ __align__(16) _Float16 w1l[64][64];        //  8 KB
  __shared__ __align__(16) _Float16 w2l[128][64];       // 16 KB
  __shared__ __align__(16) _Float16 featl[4][32][CPAD]; // 24 KB : per-wave 32x96 f16 tile

  const int tid = (int)threadIdx.x;
  const int lane = tid & 31;
  const int w = tid >> 5;

  // --- stage weights (block-cooperative, once); w0 column-permuted ---
  for (int i = tid; i < 64 * CPAD; i += 128) {
    const int o = i / CPAD, c = i % CPAD;
    float v = 0.f;
    if (c < 64)            v = w0[o * CIN + c + 3];   // points channels
    else if (c < CIN)      v = w0[o * CIN + (c - 64)]; // g_norm channels
    w0l[o][c] = (_Float16)v;
  }
  for (int i = tid; i < 64 * 64; i += 128)  w1l[i >> 6][i & 63] = (_Float16)w1[i];
  for (int i = tid; i < 128 * 64; i += 128) w2l[i >> 6][i & 63] = (_Float16)w2[i];

  const int sid = (int)blockIdx.x * 4 + w;
  const int b = sid >> 10, s = sid & 1023;
  const int n = lane & 15;

  float bias0[4], bias1[4], bias2[8];
#pragma unroll
  for (int nt = 0; nt < 4; ++nt) { bias0[nt] = b0[nt * 16 + n]; bias1[nt] = b1[nt * 16 + n]; }
#pragma unroll
  for (int nt = 0; nt < 8; ++nt) bias2[nt] = b2[nt * 16 + n];

  // --- gather: lane = one of the K=32 neighbors ---
  const int gi = idxws[sid * KK + lane];
  const float cx = new_xyz[(b * 3 + 0) * SS + s];
  const float cy = new_xyz[(b * 3 + 1) * SS + s];
  const float cz = new_xyz[(b * 3 + 2) * SS + s];

  if (ptsT != nullptr) {
    // one contiguous 128B row per point: 8 x global b128 -> 8 x ds b128
    const float4* src = reinterpret_cast<const float4*>(ptsT + ((size_t)b * NN + gi) * DD);
    float4* dst = reinterpret_cast<float4*>(&featl[w][lane][0]);
#pragma unroll
    for (int q = 0; q < 8; ++q) dst[q] = src[q];
  } else {
#pragma unroll 8
    for (int c = 0; c < DD; ++c)
      featl[w][lane][c] = (_Float16)points[((size_t)b * DD + c) * NN + gi];
  }
  featl[w][lane][64] = (_Float16)(xyz[((size_t)b * 3 + 0) * NN + gi] - cx);
  featl[w][lane][65] = (_Float16)(xyz[((size_t)b * 3 + 1) * NN + gi] - cy);
  featl[w][lane][66] = (_Float16)(xyz[((size_t)b * 3 + 2) * NN + gi] - cz);
#pragma unroll
  for (int c = CIN; c < CPAD; ++c) featl[w][lane][c] = (_Float16)0.f;

  // attention mean over the K gathered values
  float av = att[(size_t)b * NN + gi];
#pragma unroll
  for (int off = 16; off >= 1; off >>= 1) av += __shfl_xor(av, off);
  if (lane == 0) out_att[b * SS + s] = av * (1.f / 32.f);

  __syncthreads();

  float run[8];
#pragma unroll
  for (int nt = 0; nt < 8; ++nt) run[nt] = -1e30f;

  for (int mt = 0; mt < 2; ++mt) {
    _Float16* fb = &featl[w][mt * 16][0];

    // ---- layer 0: (16x96) x (96x64) ----
    v16h a0[3];
#pragma unroll
    for (int ks = 0; ks < 3; ++ks) a0[ks] = frag_a(fb, CPAD, ks * 32, lane);
    __syncthreads();  // a-loads complete before h0 overwrites the tile
#pragma unroll
    for (int nt = 0; nt < 4; ++nt) {
      v8f acc = {};
#pragma unroll
      for (int ks = 0; ks < 3; ++ks)
        acc = wmma16(a0[ks], frag_b(&w0l[0][0], CPAD, ks * 32, nt * 16, lane), acc);
      store_h(fb, CPAD, nt * 16, lane, acc, bias0[nt]);
    }
    __syncthreads();

    // ---- layer 1: (16x64) x (64x64) ----
    v16h a1[2];
#pragma unroll
    for (int ks = 0; ks < 2; ++ks) a1[ks] = frag_a(fb, CPAD, ks * 32, lane);
    __syncthreads();
#pragma unroll
    for (int nt = 0; nt < 4; ++nt) {
      v8f acc = {};
#pragma unroll
      for (int ks = 0; ks < 2; ++ks)
        acc = wmma16(a1[ks], frag_b(&w1l[0][0], 64, ks * 32, nt * 16, lane), acc);
      store_h(fb, CPAD, nt * 16, lane, acc, bias1[nt]);
    }
    __syncthreads();

    // ---- layer 2: (16x64) x (64x128), fold max over rows ----
    v16h a2[2];
#pragma unroll
    for (int ks = 0; ks < 2; ++ks) a2[ks] = frag_a(fb, CPAD, ks * 32, lane);
#pragma unroll
    for (int nt = 0; nt < 8; ++nt) {
      v8f acc = {};
#pragma unroll
      for (int ks = 0; ks < 2; ++ks)
        acc = wmma16(a2[ks], frag_b(&w2l[0][0], 64, ks * 32, nt * 16, lane), acc);
      float m = -1e30f;
#pragma unroll
      for (int g = 0; g < 8; ++g) {
        float v = acc[g] + bias2[nt];
        v = v > 0.f ? v : 0.f;
        m = fmaxf(m, v);
      }
      run[nt] = fmaxf(run[nt], m);
    }
    __syncthreads();
  }

  // combine the two lane halves (rows 0-7 vs 8-15 of each m-tile)
#pragma unroll
  for (int nt = 0; nt < 8; ++nt) run[nt] = fmaxf(run[nt], __shfl_xor(run[nt], 16));
  if (lane < 16) {
#pragma unroll
    for (int nt = 0; nt < 8; ++nt)
      out_pts[((size_t)b * 128 + nt * 16 + lane) * SS + s] = run[nt];
  }
}

extern "C" void kernel_launch(void* const* d_in, const int* in_sizes, int n_in,
                              void* d_out, int out_size, void* d_ws, size_t ws_size,
                              hipStream_t stream) {
  (void)in_sizes; (void)n_in; (void)out_size;
  const float* xyz = (const float*)d_in[0];
  const float* points = (const float*)d_in[1];
  const float* att = (const float*)d_in[2];
  const float* w0 = (const float*)d_in[3];
  const float* b0 = (const float*)d_in[4];
  const float* w1 = (const float*)d_in[5];
  const float* b1 = (const float*)d_in[6];
  const float* w2 = (const float*)d_in[7];
  const float* b2 = (const float*)d_in[8];

  float* out = (float*)d_out;
  float* out_xyz = out;                              // (B,3,S)
  float* out_pts = out + (size_t)BB * 3 * SS;        // (B,128,S)
  float* out_att = out_pts + (size_t)BB * 128 * SS;  // (B,1,S)

  const size_t idxBytes = (size_t)BB * SS * KK * sizeof(int);      // 1 MB
  const size_t ptsTBytes = (size_t)BB * NN * DD * sizeof(_Float16); // 8 MB
  int* idxws = (int*)d_ws;
  _Float16* ptsT = (ws_size >= idxBytes + ptsTBytes)
                       ? (_Float16*)((char*)d_ws + idxBytes)
                       : nullptr;

  if (ptsT) transpose_kernel<<<BB * (NN / 64), 256, 0, stream>>>(points, ptsT);
  fps_kernel<<<BB, 1024, 0, stream>>>(xyz, out_xyz);
  ballq_kernel<<<(BB * SS) / 8, 256, 0, stream>>>(xyz, out_xyz, idxws);
  mlp_kernel<<<(BB * SS) / 4, 128, 0, stream>>>(xyz, points, ptsT, att, w0, b0, w1, b1, w2, b2,
                                                idxws, out_xyz, out_pts, out_att);
}
